// GCN_EL_38714835206180
// MI455X (gfx1250) — matrix operand order repaired and checked by reading, not compiled, not run on hardware
//
#include <hip/hip_runtime.h>
#include <hip/hip_bf16.h>

typedef __attribute__((ext_vector_type(16))) __bf16 v16bf;
typedef __attribute__((ext_vector_type(8)))  __bf16 v8bf;
typedef __attribute__((ext_vector_type(8)))  float  v8f;

// ---------------------------------------------------------------------------
// WMMA helpers (CDNA5 wave32, V_WMMA_F32_16X16X32_BF16)
// ---------------------------------------------------------------------------
__device__ __forceinline__ v8f wmma_bf16(v16bf a, v16bf b, v8f c) {
    return __builtin_amdgcn_wmma_f32_16x16x32_bf16(
        /*neg_a=*/false, a, /*neg_b=*/false, b,
        /*c_mod=*/(short)0, c, /*reuse_a=*/false, /*reuse_b=*/false);
}

// A fragment: 16x32 bf16 tile from LDS, row-major with `stride` halves per row.
// Element e of lane L: M = L%16, K = kbase + (L<16?0:16) + e.
__device__ __forceinline__ v16bf load_a_frag(const __bf16* tile, int stride,
                                             int kbase, int lane) {
    const __bf16* p = tile + (lane & 15) * stride + kbase + ((lane & 16) ? 16 : 0);
    v8bf lo = *(const v8bf*)p;
    v8bf hi = *(const v8bf*)(p + 8);
    return __builtin_shufflevector(lo, hi, 0, 1, 2, 3, 4, 5, 6, 7,
                                   8, 9, 10, 11, 12, 13, 14, 15);
}

// B fragment: 32x16 bf16 from transposed weights Wt[Nout][K] in global memory.
// Element e of lane L: K = kbase + (L<16?0:16) + e, N = ntile*16 + L%16.
__device__ __forceinline__ v16bf load_b_frag(const __bf16* Wt, int K, int ntile,
                                             int kbase, int lane) {
    const __bf16* p = Wt + (size_t)(ntile * 16 + (lane & 15)) * K + kbase +
                      ((lane & 16) ? 16 : 0);
    return *(const v16bf*)p;   // 32B, 32B-aligned -> 2x global_load_b128
}

// ---------------------------------------------------------------------------
// Small prep kernels
// ---------------------------------------------------------------------------
__global__ void k_transpose_bf16(const float* __restrict__ W,
                                 __bf16* __restrict__ Wt, int K, int Nn) {
    int i = blockIdx.x * blockDim.x + threadIdx.x;
    if (i >= K * Nn) return;
    int k = i / Nn, n = i % Nn;
    Wt[n * K + k] = (__bf16)W[i];
}

// h = relu(concat(relu(x[:, :14]@We1+b1), relu(x[:,14:]@We2+b2)) @ We3 + b3)
__global__ void __launch_bounds__(128) k_node_embed(
    const float* __restrict__ x, const float* __restrict__ We1,
    const float* __restrict__ be1, const float* __restrict__ We2,
    const float* __restrict__ be2, const float* __restrict__ We3,
    const float* __restrict__ be3, __bf16* __restrict__ hb) {
    __shared__ float xr[26];
    __shared__ float hc[128];
    int n = blockIdx.x, t = threadIdx.x;
    if (t < 26) xr[t] = x[n * 26 + t];
    __syncthreads();
    float acc;
    if (t < 64) {
        acc = be1[t];
        #pragma unroll
        for (int k = 0; k < 14; ++k) acc += xr[k] * We1[k * 64 + t];
    } else {
        int o = t - 64;
        acc = be2[o];
        #pragma unroll
        for (int k = 0; k < 12; ++k) acc += xr[14 + k] * We2[k * 64 + o];
    }
    hc[t] = fmaxf(acc, 0.f);
    __syncthreads();
    float a = be3[t];
    for (int k = 0; k < 128; ++k) a += hc[k] * We3[k * 128 + t];
    hb[(size_t)n * 128 + t] = (__bf16)fmaxf(a, 0.f);
}

// e = relu(edge_attr @ Wee + bee)  -> bf16 [E,64]
__global__ void k_edge_embed(const float* __restrict__ ea,
                             const float* __restrict__ Wee,
                             const float* __restrict__ bee,
                             __bf16* __restrict__ eb, int E) {
    int i = blockIdx.x * blockDim.x + threadIdx.x;
    if (i >= E * 64) return;
    int e = i >> 6, o = i & 63;
    float acc = bee[o];
    const float* ep = ea + (size_t)e * 8;
    #pragma unroll
    for (int k = 0; k < 8; ++k) acc += ep[k] * Wee[k * 64 + o];
    eb[i] = (__bf16)fmaxf(acc, 0.f);
}

__global__ void k_deg(const int* __restrict__ dst, float* __restrict__ deg, int E) {
    int e = blockIdx.x * blockDim.x + threadIdx.x;
    if (e < E) atomicAdd(&deg[dst[e]], 1.f);
}

__global__ void k_deginv(const float* __restrict__ deg, float* __restrict__ dinv, int N) {
    int i = blockIdx.x * blockDim.x + threadIdx.x;
    if (i < N) dinv[i] = 1.f / fmaxf(deg[i], 1.f);
}

__global__ void k_cnt(const int* __restrict__ comm, float* __restrict__ cnt, int N) {
    int i = blockIdx.x * blockDim.x + threadIdx.x;
    if (i < N) atomicAdd(&cnt[comm[i]], 1.f);
}

// ---------------------------------------------------------------------------
// Core WMMA kernel: per 16-edge tile compute he, mask; fuse agg1 scatter-add
// and coalesced bf16 mask writeback. One wave per tile, 4 waves per block.
// ---------------------------------------------------------------------------
__global__ void __launch_bounds__(128) k_edge_mask(
    const __bf16* __restrict__ hb, const __bf16* __restrict__ eb,
    const __bf16* __restrict__ WtS, const __bf16* __restrict__ WtD,
    const __bf16* __restrict__ WtEa, const __bf16* __restrict__ WtM,
    const float* __restrict__ bel, const float* __restrict__ bm,
    const int* __restrict__ src, const int* __restrict__ dst,
    float* __restrict__ agg, __bf16* __restrict__ maskOut, int ntiles) {
    // per wave: aS 16x136, aD 16x136, aE 16x72, heT 16x136 (halves) = 7680
    __shared__ __align__(16) __bf16 smem[4 * 7680];
    int lane = threadIdx.x & 31;
    int wave = threadIdx.x >> 5;
    int tile = blockIdx.x * 4 + wave;
    if (tile >= ntiles) return;
    __bf16* aS = smem + wave * 7680;
    __bf16* aD = aS + 2176;
    __bf16* aE = aS + 4352;
    __bf16* hT = aS + 5504;
    int ebase = tile * 16;

    const uint4* hq = (const uint4*)hb;   // h row = 16 uint4
    const uint4* eq = (const uint4*)eb;   // e row = 8 uint4
    for (int idx = lane; idx < 256; idx += 32) {
        int row = idx >> 4, ch = idx & 15;
        int s = src[ebase + row];
        int d = dst[ebase + row];
        ((uint4*)aS)[row * 17 + ch] = hq[(size_t)s * 16 + ch];
        ((uint4*)aD)[row * 17 + ch] = hq[(size_t)d * 16 + ch];
    }
    for (int idx = lane; idx < 128; idx += 32) {
        int row = idx >> 3, ch = idx & 7;
        ((uint4*)aE)[row * 9 + ch] = eq[(size_t)(ebase + row) * 8 + ch];
    }
    asm volatile("" ::: "memory");   // same-wave DS ops are in-order

    int mof = (lane & 16) ? 8 : 0;
    int lc = lane & 15;

    // he = relu(h[src]Ws + h[dst]Wd + eWea + b)
    for (int n = 0; n < 8; ++n) {
        float b = bel[n * 16 + lc];
        v8f acc = {b, b, b, b, b, b, b, b};
        v8f acc2 = {0.f, 0.f, 0.f, 0.f, 0.f, 0.f, 0.f, 0.f};
        #pragma unroll
        for (int kc = 0; kc < 4; ++kc) {
            acc  = wmma_bf16(load_a_frag(aS, 136, kc * 32, lane),
                             load_b_frag(WtS, 128, n, kc * 32, lane), acc);
            acc2 = wmma_bf16(load_a_frag(aD, 136, kc * 32, lane),
                             load_b_frag(WtD, 128, n, kc * 32, lane), acc2);
        }
        #pragma unroll
        for (int kc = 0; kc < 2; ++kc) {
            acc = wmma_bf16(load_a_frag(aE, 72, kc * 32, lane),
                            load_b_frag(WtEa, 64, n, kc * 32, lane), acc);
        }
        acc = acc + acc2;
        #pragma unroll
        for (int r = 0; r < 8; ++r)
            hT[(r + mof) * 136 + n * 16 + lc] = (__bf16)fmaxf(acc[r], 0.f);
    }
    asm volatile("" ::: "memory");

    // mask = sigmoid(he @ Wm + bm); fused agg1 += h[src]*mask; stage mask in aD
    for (int n = 0; n < 8; ++n) {
        float b = bm[n * 16 + lc];
        v8f acc = {b, b, b, b, b, b, b, b};
        #pragma unroll
        for (int kc = 0; kc < 4; ++kc)
            acc = wmma_bf16(load_a_frag(hT, 136, kc * 32, lane),
                            load_b_frag(WtM, 128, n, kc * 32, lane), acc);
        #pragma unroll
        for (int r = 0; r < 8; ++r) {
            int M = r + mof;
            float m = 1.f / (1.f + __expf(-acc[r]));
            float hv = (float)aS[M * 136 + n * 16 + lc];
            int dn = dst[ebase + M];
            atomicAdd(agg + (size_t)dn * 128 + n * 16 + lc, hv * m);
            aD[M * 136 + n * 16 + lc] = (__bf16)m;
        }
    }
    asm volatile("" ::: "memory");
    uint4* mq = (uint4*)maskOut;
    for (int idx = lane; idx < 256; idx += 32) {
        int row = idx >> 4, ch = idx & 15;
        mq[(size_t)(ebase + row) * 16 + ch] = ((uint4*)aD)[row * 17 + ch];
    }
}

// aggs_bf16 = bf16(agg * deg_inv[row])
__global__ void k_scale(const float* __restrict__ agg, const float* __restrict__ dinv,
                        __bf16* __restrict__ aggs, int N) {
    int i = blockIdx.x * blockDim.x + threadIdx.x;
    if (i >= N * 128) return;
    aggs[i] = (__bf16)(agg[i] * dinv[i >> 7]);
}

// out = relu(A1 @ W1 + A2 @ W2 + b), 16-row tiles, WMMA. out f32 [N,128].
__global__ void __launch_bounds__(128) k_node_gemm(
    const __bf16* __restrict__ a1, const __bf16* __restrict__ Wt1,
    const __bf16* __restrict__ a2, const __bf16* __restrict__ Wt2,
    const float* __restrict__ bias, float* __restrict__ out, int ntiles) {
    __shared__ __align__(16) __bf16 smem[4 * 4352];
    int lane = threadIdx.x & 31;
    int wave = threadIdx.x >> 5;
    int tile = blockIdx.x * 4 + wave;
    if (tile >= ntiles) return;
    __bf16* t1 = smem + wave * 4352;
    __bf16* t2 = t1 + 2176;
    int rbase = tile * 16;
    const uint4* q1 = (const uint4*)a1;
    const uint4* q2 = (const uint4*)a2;
    for (int idx = lane; idx < 256; idx += 32) {
        int row = idx >> 4, ch = idx & 15;
        ((uint4*)t1)[row * 17 + ch] = q1[(size_t)(rbase + row) * 16 + ch];
        ((uint4*)t2)[row * 17 + ch] = q2[(size_t)(rbase + row) * 16 + ch];
    }
    asm volatile("" ::: "memory");
    int mof = (lane & 16) ? 8 : 0;
    int lc = lane & 15;
    for (int n = 0; n < 8; ++n) {
        float b = bias[n * 16 + lc];
        v8f acc = {b, b, b, b, b, b, b, b};
        v8f acc2 = {0.f, 0.f, 0.f, 0.f, 0.f, 0.f, 0.f, 0.f};
        #pragma unroll
        for (int kc = 0; kc < 4; ++kc) {
            acc  = wmma_bf16(load_a_frag(t1, 136, kc * 32, lane),
                             load_b_frag(Wt1, 128, n, kc * 32, lane), acc);
            acc2 = wmma_bf16(load_a_frag(t2, 136, kc * 32, lane),
                             load_b_frag(Wt2, 128, n, kc * 32, lane), acc2);
        }
        acc = acc + acc2;
        #pragma unroll
        for (int r = 0; r < 8; ++r)
            out[(size_t)(rbase + r + mof) * 128 + n * 16 + lc] = fmaxf(acc[r], 0.f);
    }
}

__global__ void k_f32_to_bf16(const float* __restrict__ a, __bf16* __restrict__ b, int n) {
    int i = blockIdx.x * blockDim.x + threadIdx.x;
    if (i < n) b[i] = (__bf16)a[i];
}

// pass 2: agg2[dst] += h1[src] * mask (elementwise, bf16 in, f32 atomics)
__global__ void k_apply_mask(const __bf16* __restrict__ h1b, const __bf16* __restrict__ mb,
                             const int* __restrict__ src, const int* __restrict__ dst,
                             float* __restrict__ agg, int E) {
    int i = blockIdx.x * blockDim.x + threadIdx.x;
    if (i >= E * 16) return;
    int e = i >> 4, c0 = (i & 15) * 8;
    int s = src[e], d = dst[e];
    v8bf hv = *(const v8bf*)(h1b + (size_t)s * 128 + c0);
    v8bf mv = *(const v8bf*)(mb + (size_t)e * 128 + c0);
    float* ag = agg + (size_t)d * 128 + c0;
    #pragma unroll
    for (int j = 0; j < 8; ++j) atomicAdd(&ag[j], (float)hv[j] * (float)mv[j]);
}

// pooling: psum += h ; pmax = max(pmax, h) via uint bit-pattern (h >= 0 post-ReLU)
__global__ void k_pool(const float* __restrict__ h, const int* __restrict__ comm,
                       float* __restrict__ psum, unsigned* __restrict__ pmax, int N) {
    int i = blockIdx.x * blockDim.x + threadIdx.x;
    if (i >= N * 32) return;
    int n = i >> 5, c0 = (i & 31) * 4;
    int c = comm[n];
    float4 v = *(const float4*)(h + (size_t)n * 128 + c0);
    float* ps = psum + (size_t)c * 128 + c0;
    unsigned* pm = pmax + (size_t)c * 128 + c0;
    atomicAdd(&ps[0], v.x); atomicAdd(&ps[1], v.y);
    atomicAdd(&ps[2], v.z); atomicAdd(&ps[3], v.w);
    atomicMax(&pm[0], __float_as_uint(v.x)); atomicMax(&pm[1], __float_as_uint(v.y));
    atomicMax(&pm[2], __float_as_uint(v.z)); atomicMax(&pm[3], __float_as_uint(v.w));
}

// out[c] = relu(p @ Wl1 + bl1) @ Wl2 + bl2, p = [psum/cnt, pmax1+pmax2]
__global__ void __launch_bounds__(128) k_final(
    const float* __restrict__ psum, const unsigned* __restrict__ pm1,
    const unsigned* __restrict__ pm2, const float* __restrict__ cnt,
    const float* __restrict__ Wl1, const float* __restrict__ bl1,
    const float* __restrict__ Wl2, const float* __restrict__ bl2,
    float* __restrict__ out) {
    __shared__ float p[256];
    __shared__ float q[128];
    int c = blockIdx.x, t = threadIdx.x;
    float ic = 1.f / fmaxf(cnt[c], 1.f);
    p[t] = psum[(size_t)c * 128 + t] * ic;
    p[128 + t] = __uint_as_float(pm1[(size_t)c * 128 + t]) +
                 __uint_as_float(pm2[(size_t)c * 128 + t]);
    __syncthreads();
    float acc = bl1[t];
    for (int k = 0; k < 256; ++k) acc += p[k] * Wl1[k * 128 + t];
    q[t] = fmaxf(acc, 0.f) * Wl2[t];
    __syncthreads();
    for (int s = 64; s > 0; s >>= 1) {
        if (t < s) q[t] += q[t + s];
        __syncthreads();
    }
    if (t == 0) out[c] = q[0] + bl2[0];
}

// ---------------------------------------------------------------------------
extern "C" void kernel_launch(void* const* d_in, const int* in_sizes, int n_in,
                              void* d_out, int out_size, void* d_ws, size_t ws_size,
                              hipStream_t stream) {
    const float* x      = (const float*)d_in[0];
    const float* eattr  = (const float*)d_in[1];
    const float* W_ee   = (const float*)d_in[2];  const float* b_ee = (const float*)d_in[3];
    const float* W_e1   = (const float*)d_in[4];  const float* b_e1 = (const float*)d_in[5];
    const float* W_e2   = (const float*)d_in[6];  const float* b_e2 = (const float*)d_in[7];
    const float* W_e3   = (const float*)d_in[8];  const float* b_e3 = (const float*)d_in[9];
    const float* W_src  = (const float*)d_in[10]; const float* W_dst = (const float*)d_in[11];
    const float* W_ea   = (const float*)d_in[12]; const float* b_el = (const float*)d_in[13];
    const float* W_m    = (const float*)d_in[14]; const float* b_m  = (const float*)d_in[15];
    const float* W_n1   = (const float*)d_in[16]; const float* W_r1 = (const float*)d_in[17];
    const float* b_c1   = (const float*)d_in[18];
    const float* W_n2   = (const float*)d_in[19]; const float* W_r2 = (const float*)d_in[20];
    const float* b_c2   = (const float*)d_in[21];
    const float* W_l1   = (const float*)d_in[22]; const float* b_l1 = (const float*)d_in[23];
    const float* W_l2   = (const float*)d_in[24]; const float* b_l2 = (const float*)d_in[25];
    const int* eidx     = (const int*)d_in[26];
    const int* comm     = (const int*)d_in[27];
    float* out = (float*)d_out;

    const int N = in_sizes[27];       // 50000
    const int E = in_sizes[1] / 8;    // 800000
    const int C = out_size;           // 2000
    const int* src = eidx;
    const int* dst = eidx + E;

    // --- workspace carve (256B-aligned) ---
    char* w = (char*)d_ws;
    size_t off = 0;
    auto take = [&](size_t bytes) -> char* {
        char* p = w + off;
        off = (off + bytes + 255) & ~(size_t)255;
        return p;
    };
    __bf16* WT_S  = (__bf16*)take(128 * 128 * 2);
    __bf16* WT_D  = (__bf16*)take(128 * 128 * 2);
    __bf16* WT_EA = (__bf16*)take(128 * 64 * 2);
    __bf16* WT_M  = (__bf16*)take(128 * 128 * 2);
    __bf16* WT_N1 = (__bf16*)take(128 * 128 * 2);
    __bf16* WT_R1 = (__bf16*)take(128 * 128 * 2);
    __bf16* WT_N2 = (__bf16*)take(128 * 128 * 2);
    __bf16* WT_R2 = (__bf16*)take(128 * 128 * 2);
    __bf16* HB    = (__bf16*)take((size_t)N * 128 * 2);
    __bf16* EB    = (__bf16*)take((size_t)E * 64 * 2);
    __bf16* MB    = (__bf16*)take((size_t)E * 128 * 2);
    float*  AGG   = (float*)take((size_t)N * 128 * 4);
    __bf16* AGGS  = (__bf16*)take((size_t)N * 128 * 2);
    float*  H1F   = (float*)take((size_t)N * 128 * 4);
    __bf16* H1B   = (__bf16*)take((size_t)N * 128 * 2);
    float*  H2F   = (float*)take((size_t)N * 128 * 4);
    float*  DEG   = (float*)take((size_t)N * 4);
    float*  DINV  = (float*)take((size_t)N * 4);
    float*  CNT   = (float*)take((size_t)C * 4);
    float*  PSUM  = (float*)take((size_t)C * 128 * 4);
    unsigned* PM1 = (unsigned*)take((size_t)C * 128 * 4);
    unsigned* PM2 = (unsigned*)take((size_t)C * 128 * 4);

    // --- zero-init accumulators (graph-capturable stream memsets) ---
    hipMemsetAsync(AGG, 0, (size_t)N * 128 * 4, stream);
    hipMemsetAsync(DEG, 0, (size_t)N * 4, stream);
    hipMemsetAsync(CNT, 0, (size_t)C * 4, stream);
    hipMemsetAsync(PSUM, 0, (size_t)C * 128 * 4, stream);
    hipMemsetAsync(PM1, 0, (size_t)C * 128 * 4, stream);
    hipMemsetAsync(PM2, 0, (size_t)C * 128 * 4, stream);

    // --- weight transpose + bf16 convert ---
    k_transpose_bf16<<<64, 256, 0, stream>>>(W_src, WT_S, 128, 128);
    k_transpose_bf16<<<64, 256, 0, stream>>>(W_dst, WT_D, 128, 128);
    k_transpose_bf16<<<32, 256, 0, stream>>>(W_ea, WT_EA, 64, 128);
    k_transpose_bf16<<<64, 256, 0, stream>>>(W_m, WT_M, 128, 128);
    k_transpose_bf16<<<64, 256, 0, stream>>>(W_n1, WT_N1, 128, 128);
    k_transpose_bf16<<<64, 256, 0, stream>>>(W_r1, WT_R1, 128, 128);
    k_transpose_bf16<<<64, 256, 0, stream>>>(W_n2, WT_N2, 128, 128);
    k_transpose_bf16<<<64, 256, 0, stream>>>(W_r2, WT_R2, 128, 128);

    // --- embeddings, degree, counts ---
    k_node_embed<<<N, 128, 0, stream>>>(x, W_e1, b_e1, W_e2, b_e2, W_e3, b_e3, HB);
    k_edge_embed<<<(E * 64 + 255) / 256, 256, 0, stream>>>(eattr, W_ee, b_ee, EB, E);
    k_deg<<<(E + 255) / 256, 256, 0, stream>>>(dst, DEG, E);
    k_deginv<<<(N + 255) / 256, 256, 0, stream>>>(DEG, DINV, N);
    k_cnt<<<(N + 255) / 256, 256, 0, stream>>>(comm, CNT, N);

    // --- edge WMMA: he + mask + fused agg1 scatter ---
    int etiles = (E + 15) / 16;                       // 50000
    k_edge_mask<<<(etiles + 3) / 4, 128, 0, stream>>>(
        HB, EB, WT_S, WT_D, WT_EA, WT_M, b_el, b_m, src, dst, AGG, MB, etiles);

    // --- conv1: h1 = relu(agg1s @ Wn1 + h @ Wr1 + bc1) ---
    k_scale<<<(N * 128 + 255) / 256, 256, 0, stream>>>(AGG, DINV, AGGS, N);
    int ntiles = (N + 15) / 16;                       // 3125
    k_node_gemm<<<(ntiles + 3) / 4, 128, 0, stream>>>(AGGS, WT_N1, HB, WT_R1, b_c1, H1F, ntiles);
    k_f32_to_bf16<<<(N * 128 + 255) / 256, 256, 0, stream>>>(H1F, H1B, N * 128);
    k_pool<<<(N * 32 + 255) / 256, 256, 0, stream>>>(H1F, comm, PSUM, PM1, N);

    // --- conv2: agg2 from stored mask, then h2 ---
    hipMemsetAsync(AGG, 0, (size_t)N * 128 * 4, stream);
    k_apply_mask<<<(E * 16 + 255) / 256, 256, 0, stream>>>(H1B, MB, src, dst, AGG, E);
    k_scale<<<(N * 128 + 255) / 256, 256, 0, stream>>>(AGG, DINV, AGGS, N);
    k_node_gemm<<<(ntiles + 3) / 4, 128, 0, stream>>>(AGGS, WT_N2, H1B, WT_R2, b_c2, H2F, ntiles);
    k_pool<<<(N * 32 + 255) / 256, 256, 0, stream>>>(H2F, comm, PSUM, PM2, N);

    // --- readout MLP per community ---
    k_final<<<C, 128, 0, stream>>>(PSUM, PM1, PM2, CNT, W_l1, b_l1, W_l2, b_l2, out);
}
